// MultiHeadAttention_26895085207631
// MI455X (gfx1250) — matrix operand order repaired and checked by reading, not compiled, not run on hardware
//
#include <hip/hip_runtime.h>

// ---------------- types ----------------
typedef __attribute__((ext_vector_type(16))) __bf16       v16bf;
typedef __attribute__((ext_vector_type(8)))  float        v8f;
typedef __attribute__((ext_vector_type(4))) unsigned int  u32x4;
typedef __attribute__((ext_vector_type(8))) unsigned int  u32x8;
typedef __attribute__((ext_vector_type(4)))  float        f32x4;

struct U32x4x2 { u32x4 a, b; };

__device__ __forceinline__ v16bf pack16(u32x4 a, u32x4 b) {
  U32x4x2 t{a, b};
  return __builtin_bit_cast(v16bf, t);
}

// B-fragment: col = lane%16, elems 0-7 at p, elems 8-15 at p+16 (p = colbase + k0 + (lane/16)*8)
__device__ __forceinline__ v16bf load16(const __bf16* p) {
  return pack16(*(const u32x4*)p, *(const u32x4*)(p + 16));
}
// A-fragment: elems 0-7 at p, elems 8-15 at p+16 (p = rowbase + k0 + (lane/16)*8)
__device__ __forceinline__ v16bf loadA(const __bf16* p) {
  return pack16(*(const u32x4*)p, *(const u32x4*)(p + 16));
}

__device__ __forceinline__ v8f vzero() {
  v8f z = {0.f, 0.f, 0.f, 0.f, 0.f, 0.f, 0.f, 0.f};
  return z;
}

__device__ __forceinline__ v8f wmma_bf16(v16bf a, v16bf b, v8f c) {
  return __builtin_amdgcn_wmma_f32_16x16x32_bf16(false, a, false, b, (short)0, c,
                                                 false, false);
}

__device__ __forceinline__ float redmax16(float v) {
  v = fmaxf(v, __shfl_xor(v, 1, 32));
  v = fmaxf(v, __shfl_xor(v, 2, 32));
  v = fmaxf(v, __shfl_xor(v, 4, 32));
  v = fmaxf(v, __shfl_xor(v, 8, 32));
  return v;
}
__device__ __forceinline__ float redsum16(float v) {
  v += __shfl_xor(v, 1, 32);
  v += __shfl_xor(v, 2, 32);
  v += __shfl_xor(v, 4, 32);
  v += __shfl_xor(v, 8, 32);
  return v;
}

#define SEQ   2048
#define DM    512
#define DK    64
#define NEG_BIG (-3.0e38f)

// ---------------- TDM: 2D tensor tile -> LDS (CDNA5 Tensor Data Mover) ----------------
// D# per CDNA5 ISA. data_size=1 (2-byte bf16).
__device__ __forceinline__ void tdm_load_2d_bf16(unsigned lds_addr,
                                                 unsigned long long gaddr,
                                                 unsigned tile_d0, unsigned tile_d1,
                                                 unsigned long long tensor_d0,
                                                 unsigned long long tensor_d1,
                                                 unsigned long long stride0) {
  u32x4 g0;
  g0[0] = 1u;                                              // count=1 (user mode)
  g0[1] = lds_addr;                                        // LDS byte address
  g0[2] = (unsigned)(gaddr & 0xffffffffull);               // global_addr[31:0]
  g0[3] = (unsigned)((gaddr >> 32) & 0x1ffffffull) | (2u << 30);  // [56:32] | type=2
  u32x8 g1;
  g1[0] = (1u << 16);                                      // data_size=1 (2B)
  g1[1] = ((unsigned)(tensor_d0 & 0xffffull)) << 16;       // tensor_dim0[15:0]
  g1[2] = (unsigned)((tensor_d0 >> 16) & 0xffffull) |
          (((unsigned)(tensor_d1 & 0xffffull)) << 16);     // t_d0[31:16] | t_d1[15:0]
  g1[3] = (unsigned)((tensor_d1 >> 16) & 0xffffull) |
          ((tile_d0 & 0xffffu) << 16);                     // t_d1[31:16] | tile_dim0
  g1[4] = tile_d1 & 0xffffu;                               // tile_dim1 ; tile_dim2=0
  g1[5] = (unsigned)(stride0 & 0xffffffffull);             // dim0_stride[31:0]
  g1[6] = (unsigned)((stride0 >> 32) & 0xffffull);         // dim0_stride[47:32]
  g1[7] = 0u;                                              // dim1_stride=0 (2D)
  u32x4 gz = {0u, 0u, 0u, 0u};
  asm volatile("tensor_load_to_lds %0, %1, %2, %3"
               :: "s"(g0), "s"(g1), "s"(gz), "s"(gz)
               : "memory");
}

// ---------------- kernel 1: weight prep ----------------
__global__ void prep_kernel(const float* __restrict__ WQ, const float* __restrict__ WK,
                            const float* __restrict__ WV, const float* __restrict__ Wo,
                            __bf16* __restrict__ WTq, __bf16* __restrict__ WTk,
                            __bf16* __restrict__ WTv, __bf16* __restrict__ WoT) {
  int idx = blockIdx.x * blockDim.x + threadIdx.x;  // 0 .. 131071
  if (idx < 3 * 32768) {
    int m = idx >> 15;
    int r = idx & 32767;
    int d = r >> 9;   // 0..63
    int c = r & 511;  // 0..511
    const float* W = (m == 0) ? WQ : (m == 1) ? WK : WV;
    __bf16* WT = (m == 0) ? WTq : (m == 1) ? WTk : WTv;
    WT[d * 512 + c] = (__bf16)W[c * 64 + d];
  } else {
    int r = idx - 3 * 32768;  // 0..32767
    int n = r >> 6;           // out col 0..511
    int k = r & 63;           // dk row 0..63
    float s = 0.f;
#pragma unroll
    for (int h = 0; h < 8; ++h) s += Wo[(h * 64 + k) * 512 + n];
    WoT[n * 64 + k] = (__bf16)s;
  }
}

// ---------------- kernel 2: fused QKV projection (WMMA) ----------------
__global__ __launch_bounds__(32)
void proj_kernel(const float* __restrict__ Q, const float* __restrict__ K,
                 const float* __restrict__ V,
                 const __bf16* __restrict__ WTq, const __bf16* __restrict__ WTk,
                 const __bf16* __restrict__ WTv,
                 const float* __restrict__ bq, const float* __restrict__ bk,
                 const float* __restrict__ bv,
                 __bf16* __restrict__ qh, __bf16* __restrict__ kh,
                 __bf16* __restrict__ vT) {
  const int wid = blockIdx.x;  // 0..1535
  const int mat = wid / 512;   // 0=q 1=k 2=v
  const int rt  = wid % 512;   // row tile over B*S
  const int lane = threadIdx.x & 31;
  const int l16 = lane & 15, half = lane >> 4, sub8 = half * 8;

  const float*  In   = (mat == 0) ? Q   : (mat == 1) ? K   : V;
  const __bf16* WT   = (mat == 0) ? WTq : (mat == 1) ? WTk : WTv;
  const float*  bias = (mat == 0) ? bq  : (mat == 1) ? bk  : bv;

  const float* arow = In + (size_t)(rt * 16 + l16) * DM;

  v8f acc[4];
#pragma unroll
  for (int t = 0; t < 4; ++t) acc[t] = vzero();

  for (int kk = 0; kk < DM; kk += 32) {
    f32x4 x0 = *(const f32x4*)(arow + kk + sub8);
    f32x4 x1 = *(const f32x4*)(arow + kk + sub8 + 4);
    f32x4 x2 = *(const f32x4*)(arow + kk + 16 + sub8);
    f32x4 x3 = *(const f32x4*)(arow + kk + 16 + sub8 + 4);
    float fa[16];
#pragma unroll
    for (int j = 0; j < 4; ++j) {
      fa[j] = x0[j]; fa[4 + j] = x1[j]; fa[8 + j] = x2[j]; fa[12 + j] = x3[j];
    }
    v16bf a;
#pragma unroll
    for (int j = 0; j < 16; ++j) a[j] = (__bf16)fa[j];

#pragma unroll
    for (int t = 0; t < 4; ++t) {
      v16bf b = load16(WT + (t * 16 + l16) * DM + kk + half * 8);
      acc[t] = wmma_bf16(a, b, acc[t]);
    }
  }

  const int bidx = rt >> 7;
  const int s0   = (rt & 127) * 16;
#pragma unroll
  for (int t = 0; t < 4; ++t) {
    const int d = t * 16 + l16;
    const float bb = bias[d];
#pragma unroll
    for (int r = 0; r < 8; ++r) {
      float val = acc[t][r] + bb;
      if (mat == 0) val *= 0.125f;  // 1/sqrt(dk)
      __bf16 bf = (__bf16)val;
      const int row = rt * 16 + sub8 + r;
      if (mat == 0)      qh[row * DK + d] = bf;
      else if (mat == 1) kh[row * DK + d] = bf;
      else               vT[((size_t)bidx * DK + d) * SEQ + s0 + sub8 + r] = bf;
    }
  }
}

// ---------------- kernel 3: flash attention + fused output projection ----------------
// Block = 4 waves = 64 query rows. K/V tiles (64 keys x 64 dk) staged to LDS by the
// Tensor Data Mover with DOUBLE BUFFERING: wave 0 issues block i+1's DMA before
// waiting on block i (in-order TDM completion => s_wait_tensorcnt(2) proves block i
// landed), so DMA streams behind compute instead of serializing at the barrier.
__global__ __launch_bounds__(128)
void attn_kernel(const __bf16* __restrict__ qh, const __bf16* __restrict__ kh,
                 const __bf16* __restrict__ vT, const __bf16* __restrict__ WoT,
                 const float* __restrict__ wob, float* __restrict__ out) {
  __shared__ __align__(16) __bf16 ldsK[2][64 * 64];  // [buf][key][dk]
  __shared__ __align__(16) __bf16 ldsV[2][64 * 64];  // [buf][dk][key]
  __shared__ __align__(16) __bf16 ldsP[4 * 1024];    // per-wave scratch
  const int wave = threadIdx.x >> 5;
  const int lane = threadIdx.x & 31;
  const int l16 = lane & 15, half = lane >> 4, sub8 = half * 8;

  const int blk = blockIdx.x;       // 0..127
  const int b   = blk >> 5;         // 32 blocks per batch
  const int qb  = (blk & 31) * 64;  // block's first query row within batch
  const int qs0 = qb + wave * 16;   // wave's first query row

  __bf16* myl = ldsP + wave * 1024;

  const __bf16* qrow = qh + (size_t)(b * SEQ + qs0 + l16) * DK;
  const v16bf aq0 = loadA(qrow + 0  + sub8);
  const v16bf aq1 = loadA(qrow + 32 + sub8);

  const unsigned long long kg =
      (unsigned long long)(size_t)(kh + (size_t)(b * SEQ) * DK);
  const unsigned long long vg =
      (unsigned long long)(size_t)(vT + (size_t)b * DK * SEQ);
  const unsigned ldsKo[2] = {(unsigned)(size_t)(void*)&ldsK[0][0],
                             (unsigned)(size_t)(void*)&ldsK[1][0]};
  const unsigned ldsVo[2] = {(unsigned)(size_t)(void*)&ldsV[0][0],
                             (unsigned)(size_t)(void*)&ldsV[1][0]};

  auto issue_block = [&](int i, int buf) {
    const int kb = i * 64;
    // K tile: 64 rows (keys) x 64 dk, row stride 64
    tdm_load_2d_bf16(ldsKo[buf], kg + (unsigned long long)kb * DK * 2,
                     DK, 64, DK, (unsigned long long)(SEQ - kb), DK);
    // V tile: 64 rows (dk) x 64 keys, row stride SEQ
    tdm_load_2d_bf16(ldsVo[buf], vg + (unsigned long long)kb * 2,
                     64, DK, (unsigned long long)(SEQ - kb), DK, SEQ);
  };

  float m[8], l[8];
  v8f O[4];
#pragma unroll
  for (int r = 0; r < 8; ++r) { m[r] = NEG_BIG; l[r] = 0.f; }
#pragma unroll
  for (int t = 0; t < 4; ++t) O[t] = vzero();

  const int nkb = (qb >> 6) + 1;  // uniform trip count per block
  if (wave == 0) issue_block(0, 0);

  for (int i = 0; i < nkb; ++i) {
    const int kb  = i * 64;
    const int buf = i & 1;
    if (wave == 0) {
      if (i + 1 < nkb) {
        issue_block(i + 1, buf ^ 1);          // prefetch next block (other buffer)
        __builtin_amdgcn_s_wait_tensorcnt(2); // block i's pair has completed
      } else {
        __builtin_amdgcn_s_wait_tensorcnt(0); // last block: drain
      }
    }
    __syncthreads();  // tiles for block i visible to all waves
    const __bf16* Kt = &ldsK[buf][0];
    const __bf16* Vt = &ldsV[buf][0];

#pragma unroll
    for (int g = 0; g < 2; ++g) {
      const int kb2 = kb + g * 32;
      if (kb2 < qs0 + 16) {  // wave-uniform: any valid key in this 32-group
        v8f S[2];
#pragma unroll
        for (int t = 0; t < 2; ++t) {
          const int kt = kb2 + t * 16;
          v8f s = vzero();
          if (kt <= qs0 + 15) {
            const __bf16* krow = Kt + (kt - kb + l16) * 64;
            s = wmma_bf16(aq0, load16(krow + half * 8), s);
            s = wmma_bf16(aq1, load16(krow + 32 + half * 8), s);
#pragma unroll
            for (int r = 0; r < 8; ++r)
              if (kt + l16 > qs0 + sub8 + r) s[r] = NEG_BIG;  // causal mask
          } else {
#pragma unroll
            for (int r = 0; r < 8; ++r) s[r] = NEG_BIG;
          }
          S[t] = s;
        }

        float alpha[8], p0[8], p1[8];
#pragma unroll
        for (int r = 0; r < 8; ++r) {
          float mx = redmax16(fmaxf(S[0][r], S[1][r]));
          float mn = fmaxf(m[r], mx);
          alpha[r] = __expf(m[r] - mn);
          m[r] = mn;
        }
#pragma unroll
        for (int r = 0; r < 8; ++r) {
          p0[r] = __expf(S[0][r] - m[r]);
          p1[r] = __expf(S[1][r] - m[r]);
          l[r] = l[r] * alpha[r] + redsum16(p0[r] + p1[r]);
        }
#pragma unroll
        for (int t = 0; t < 4; ++t)
#pragma unroll
          for (int r = 0; r < 8; ++r) O[t][r] *= alpha[r];

        // P: C-layout -> LDS -> A-layout (wave-private region; DS in-order)
#pragma unroll
        for (int r = 0; r < 8; ++r) {
          myl[(sub8 + r) * 32 + l16]      = (__bf16)p0[r];
          myl[(sub8 + r) * 32 + 16 + l16] = (__bf16)p1[r];
        }
        asm volatile("" ::: "memory");
        v16bf aP = loadA(myl + l16 * 32 + sub8);

#pragma unroll
        for (int t = 0; t < 4; ++t) {
          v16bf bv = load16(Vt + (t * 16 + l16) * 64 + g * 32 + half * 8);
          O[t] = wmma_bf16(aP, bv, O[t]);
        }
      }
    }
    __syncthreads();  // all waves done with buffer before its next TDM overwrite
  }

  // normalize
#pragma unroll
  for (int r = 0; r < 8; ++r) {
    float inv = 1.f / l[r];
#pragma unroll
    for (int t = 0; t < 4; ++t) O[t][r] *= inv;
  }

  // head_out -> LDS [16][64] -> A-frags, then @ WoT + bias
#pragma unroll
  for (int t = 0; t < 4; ++t)
#pragma unroll
    for (int r = 0; r < 8; ++r)
      myl[(sub8 + r) * 64 + t * 16 + l16] = (__bf16)O[t][r];
  asm volatile("" ::: "memory");
  const __bf16* orow = myl + l16 * 64;
  v16bf aO0 = loadA(orow + 0  + sub8);
  v16bf aO1 = loadA(orow + 32 + sub8);

  float* obase = out + (size_t)(b * SEQ + qs0) * DM;
  for (int nt = 0; nt < 32; ++nt) {
    const int n = nt * 16 + l16;
    const __bf16* wrow = WoT + n * DK;
    v8f c = vzero();
    c = wmma_bf16(aO0, load16(wrow + half * 8), c);
    c = wmma_bf16(aO1, load16(wrow + 32 + half * 8), c);
    const float bb = wob[n];
#pragma unroll
    for (int r = 0; r < 8; ++r)
      obase[(size_t)(sub8 + r) * DM + n] = c[r] + bb;
  }
}

// ---------------- launch ----------------
extern "C" void kernel_launch(void* const* d_in, const int* in_sizes, int n_in,
                              void* d_out, int out_size, void* d_ws, size_t ws_size,
                              hipStream_t stream) {
  (void)in_sizes; (void)n_in; (void)out_size; (void)ws_size;
  const float* Q    = (const float*)d_in[0];
  const float* K    = (const float*)d_in[1];
  const float* V    = (const float*)d_in[2];
  const float* WQ_w = (const float*)d_in[3];
  const float* WQ_b = (const float*)d_in[4];
  const float* WK_w = (const float*)d_in[5];
  const float* WK_b = (const float*)d_in[6];
  const float* WV_w = (const float*)d_in[7];
  const float* WV_b = (const float*)d_in[8];
  const float* Wo_w = (const float*)d_in[9];
  const float* Wo_b = (const float*)d_in[10];

  char* ws = (char*)d_ws;
  const size_t SZ_QKV = (size_t)4 * SEQ * DK * sizeof(__bf16);  // 1 MB each
  const size_t SZ_WT  = (size_t)DK * DM * sizeof(__bf16);       // 64 KB each
  __bf16* qh  = (__bf16*)(ws);
  __bf16* kh  = (__bf16*)(ws + SZ_QKV);
  __bf16* vT  = (__bf16*)(ws + 2 * SZ_QKV);
  __bf16* WTq = (__bf16*)(ws + 3 * SZ_QKV);
  __bf16* WTk = (__bf16*)(ws + 3 * SZ_QKV + SZ_WT);
  __bf16* WTv = (__bf16*)(ws + 3 * SZ_QKV + 2 * SZ_WT);
  __bf16* WoT = (__bf16*)(ws + 3 * SZ_QKV + 3 * SZ_WT);

  prep_kernel<<<512, 256, 0, stream>>>(WQ_w, WK_w, WV_w, Wo_w, WTq, WTk, WTv, WoT);
  proj_kernel<<<1536, 32, 0, stream>>>(Q, K, V, WTq, WTk, WTv, WQ_b, WK_b, WV_b,
                                       qh, kh, vT);
  attn_kernel<<<128, 128, 0, stream>>>(qh, kh, vT, WoT, Wo_b, (float*)d_out);
}
